// Attention_66305705116068
// MI455X (gfx1250) — matrix operand order repaired and checked by reading, not compiled, not run on hardware
//
#include <hip/hip_runtime.h>

typedef __attribute__((ext_vector_type(16))) __bf16 v16bf;
typedef __attribute__((ext_vector_type(8)))  float  v8f;

// ---------------- constants ----------------
#define BATCH   2
#define CH      256
#define HW      2304      // 48*48
#define NHEADS  8
#define DH      64
#define HID     512       // 8*64
#define NKV     2308      // 4 mem tokens + 2304
#define NKVP    2336      // padded to 73 * 32
#define MEMTOK  4

// ---------------- helpers ----------------
__device__ inline unsigned short f2bf(float f) {
    union { float f; unsigned u; } v; v.f = f;
    unsigned u = v.u;
    u += 0x7FFFu + ((u >> 16) & 1u);          // round-to-nearest-even
    return (unsigned short)(u >> 16);
}

// B-fragment: 16 contiguous bf16 (one 32B chunk per lane)
__device__ inline v16bf ld_frag16(const unsigned short* p) {
    union { v16bf v; unsigned short s[16]; } u;
#pragma unroll
    for (int i = 0; i < 16; ++i) u.s[i] = p[i];
    return u.v;
}

// A-fragment: two contiguous 8-element (16B) chunks per lane
__device__ inline v16bf ld_frag8x2(const unsigned short* p0, const unsigned short* p1) {
    union { v16bf v; unsigned short s[16]; } u;
#pragma unroll
    for (int i = 0; i < 8; ++i) { u.s[i] = p0[i]; u.s[8 + i] = p1[i]; }
    return u.v;
}

__device__ inline v8f wmma_bf(v16bf a, v16bf b, v8f c) {
    return __builtin_amdgcn_wmma_f32_16x16x32_bf16(false, a, false, b, (short)0, c,
                                                   false, false);
}

// ---------------- kernel 1: weights fp32 -> bf16 ----------------
__global__ void k_convert_weights(const float* __restrict__ w_qkv,
                                  const float* __restrict__ w_out,
                                  unsigned short* __restrict__ wqkv_bf,
                                  unsigned short* __restrict__ wout_bf) {
    const int NQKV = 3 * HID * CH;       // 393216
    const int NOUT = CH * HID;           // 131072
    int idx = blockIdx.x * blockDim.x + threadIdx.x;
    if (idx < NQKV)                wqkv_bf[idx] = f2bf(w_qkv[idx]);
    else if (idx < NQKV + NOUT)    wout_bf[idx - NQKV] = f2bf(w_out[idx - NQKV]);
}

// ---------------- kernel 2: mem tokens + pad zeros into k/v workspaces ----
// k_ws layout: [b][h][j][d]   (j padded to NKVP)
// v_ws layout: [b][h][d][j]   (transposed, j padded to NKVP)
__global__ void k_prep_kv(const float* __restrict__ mem_kv,
                          unsigned short* __restrict__ k_ws,
                          unsigned short* __restrict__ v_ws) {
    int idx = blockIdx.x * blockDim.x + threadIdx.x;
    const int NMEM = BATCH * NHEADS * MEMTOK * DH;          // 4096
    const int NPAD = BATCH * NHEADS * (NKVP - NKV) * DH;    // 28672
    if (idx < NMEM) {
        int d = idx & 63;
        int j = (idx >> 6) & 3;
        int h = (idx >> 8) & 7;
        int b = idx >> 11;
        float mk = mem_kv[((0 * NHEADS + h) * MEMTOK + j) * DH + d];
        float mv = mem_kv[((1 * NHEADS + h) * MEMTOK + j) * DH + d];
        int bh = b * NHEADS + h;
        k_ws[((size_t)bh * NKVP + j) * DH + d]  = f2bf(mk);
        v_ws[((size_t)bh * DH + d) * NKVP + j]  = f2bf(mv);
    } else if (idx < NMEM + NPAD) {
        int i2 = idx - NMEM;
        int d  = i2 & 63;
        int jj = (i2 >> 6) % (NKVP - NKV);
        int bh = i2 / (64 * (NKVP - NKV));
        int j  = NKV + jj;
        k_ws[((size_t)bh * NKVP + j) * DH + d] = 0;
        v_ws[((size_t)bh * DH + d) * NKVP + j] = 0;
    }
}

// ---------------- kernel 3: RMSNorm -> bf16, layout [b][p][c] -------------
__global__ void k_rmsnorm(const float* __restrict__ x,
                          const float* __restrict__ gamma,
                          unsigned short* __restrict__ xn) {
    __shared__ float sred[CH];
    int bp = blockIdx.x;                 // 0 .. B*HW-1
    int b  = bp / HW;
    int p  = bp % HW;
    int c  = threadIdx.x;
    float val = x[((size_t)(b * CH + c)) * HW + p];
    sred[c] = val * val;
    __syncthreads();
#pragma unroll
    for (int s = CH / 2; s > 0; s >>= 1) {
        if (c < s) sred[c] += sred[c + s];
        __syncthreads();
    }
    float l2  = sqrtf(sred[0]);
    float inv = 16.0f / fmaxf(l2, 1e-12f);           // sqrt(256)=16
    xn[((size_t)bp) * CH + c] = f2bf(val * inv * (gamma[c] + 1.0f));
}

// ---------------- kernel 4: QKV GEMM (WMMA, 16x64 per wave) ---------------
// C[o,p] = sum_c Wqkv[o,c] * Xn[b,p,c];  o = part*512 + h*64 + d
// A fragment (weights) loaded once per K-step, reused across 4 N tiles.
// stores: q_ws [b][h][p][d], k_ws [b][h][p+4][d], v_ws [b][h][d][p+4]
__global__ void k_gemm_qkv(const unsigned short* __restrict__ wqkv_bf,
                           const unsigned short* __restrict__ xn,
                           unsigned short* __restrict__ q_ws,
                           unsigned short* __restrict__ k_ws,
                           unsigned short* __restrict__ v_ws) {
    const int NB = HW / 64;              // 36 column blocks of 64
    const int TM = (3 * HID) / 16;       // 96
    int bid = blockIdx.x;
    int nb  = bid % NB;
    int tm  = (bid / NB) % TM;
    int b   = bid / (NB * TM);
    int lane = threadIdx.x & 31;
    int ln = lane & 15, lh = lane >> 4;

    v8f acc[4] = {};
#pragma unroll
    for (int kc = 0; kc < CH / 32; ++kc) {
        const unsigned short* Ab = wqkv_bf + (size_t)(tm * 16 + ln) * CH + kc * 32;
        v16bf a = ld_frag8x2(Ab + 16 * lh, Ab + 8 + 16 * lh);
#pragma unroll
        for (int j = 0; j < 4; ++j) {
            const unsigned short* Bb =
                xn + ((size_t)b * HW + nb * 64 + j * 16 + ln) * CH + kc * 32 + 16 * lh;
            v16bf bm = ld_frag16(Bb);
            acc[j] = wmma_bf(a, bm, acc[j]);
        }
    }
#pragma unroll
    for (int j = 0; j < 4; ++j)
#pragma unroll
        for (int r = 0; r < 8; ++r) {
            int o    = tm * 16 + 8 * lh + r;
            int part = o >> 9;
            int rem  = o & 511;
            int h    = rem >> 6;
            int d    = rem & 63;
            int p    = nb * 64 + j * 16 + ln;
            int bh   = b * NHEADS + h;
            unsigned short hv = f2bf(acc[j][r]);
            if (part == 0)       q_ws[((size_t)bh * HW + p) * DH + d]              = hv;
            else if (part == 1)  k_ws[((size_t)bh * NKVP + (p + MEMTOK)) * DH + d] = hv;
            else                 v_ws[((size_t)bh * DH + d) * NKVP + (p + MEMTOK)] = hv;
        }
}

// ---------------- kernel 5: flash attention (WMMA, 32 queries/wave) -------
// transposed-S trick: S^T = K*Q^T so softmax row stats live on a lane pair;
// K/V A-fragments are shared across both 16-query tiles.
__global__ void k_attention(const unsigned short* __restrict__ q_ws,
                            const unsigned short* __restrict__ k_ws,
                            const unsigned short* __restrict__ v_ws,
                            unsigned short* __restrict__ y_ws) {
    __shared__ unsigned short pbuf[2 * 16 * 32];   // per u: P[query][key-in-block]
    const int QB = HW / 32;                        // 72 query blocks of 32
    int bid = blockIdx.x;
    int qb  = bid % QB;
    int h   = (bid / QB) % NHEADS;
    int b   = bid / (QB * NHEADS);
    int bh  = b * NHEADS + h;
    int qbase = qb * 32;
    int lane = threadIdx.x & 31;
    int ln = lane & 15, lh = lane >> 4;

    // Q^T B-fragments (K = d, N = queries): [query tile u][32-wide d chunk]
    v16bf qf[2][2];
#pragma unroll
    for (int u = 0; u < 2; ++u)
#pragma unroll
        for (int kk = 0; kk < 2; ++kk)
            qf[u][kk] = ld_frag16(q_ws + ((size_t)bh * HW + qbase + u * 16 + ln) * DH
                                       + 32 * kk + 16 * lh);

    v8f OT[2][4] = {};                             // O^T accum per query tile
    float m_run[2] = {-1e30f, -1e30f};
    float l_run[2] = {0.0f, 0.0f};

    for (int kb = 0; kb < NKVP; kb += 32) {
        // S^T = K * Q^T : K A-fragments shared across both query tiles
        v8f st[2][2];
#pragma unroll
        for (int t = 0; t < 2; ++t) {
            const unsigned short* ab0 =
                k_ws + ((size_t)bh * NKVP + kb + 16 * t + ln) * DH;
            v16bf ka0 = ld_frag8x2(ab0 + 16 * lh, ab0 + 8 + 16 * lh);
            v16bf ka1 = ld_frag8x2(ab0 + 32 + 16 * lh, ab0 + 40 + 16 * lh);
#pragma unroll
            for (int u = 0; u < 2; ++u) {
                v8f c = {};
                c = wmma_bf(ka0, qf[u][0], c);
                c = wmma_bf(ka1, qf[u][1], c);
                st[u][t] = c;
            }
        }

        __syncthreads();   // previous block's P reads are done
#pragma unroll
        for (int u = 0; u < 2; ++u) {
            // scale + tail mask; per-lane: 16 keys of ONE query column
            float sv[16];
            float smax = -1e30f;
#pragma unroll
            for (int t = 0; t < 2; ++t)
#pragma unroll
                for (int r = 0; r < 8; ++r) {
                    int j = kb + 16 * t + 8 * lh + r;
                    float s = (j < NKV) ? st[u][t][r] * 0.125f : -1e30f;
                    sv[t * 8 + r] = s;
                    smax = fmaxf(smax, s);
                }
            smax = fmaxf(smax, __shfl_xor(smax, 16, 32));   // lane pair = full row
            float newm = fmaxf(m_run[u], smax);
            float corr = expf(m_run[u] - newm);
            float lsum = 0.0f;
#pragma unroll
            for (int t = 0; t < 2; ++t)
#pragma unroll
                for (int r = 0; r < 8; ++r) {
                    float pp = expf(sv[t * 8 + r] - newm);
                    lsum += pp;
                    pbuf[u * 512 + ln * 32 + 16 * t + 8 * lh + r] = f2bf(pp);
                }
            lsum += __shfl_xor(lsum, 16, 32);
            l_run[u] = l_run[u] * corr + lsum;
            m_run[u] = newm;
#pragma unroll
            for (int dc = 0; dc < 4; ++dc)
#pragma unroll
                for (int r = 0; r < 8; ++r) OT[u][dc][r] *= corr;
        }
        __syncthreads();

        // P^T B-fragments for both query tiles
        v16bf pf[2];
#pragma unroll
        for (int u = 0; u < 2; ++u)
            pf[u] = ld_frag16(pbuf + u * 512 + ln * 32 + 16 * lh);

        // O^T += V^T * P^T : V A-fragments shared across both query tiles
#pragma unroll
        for (int dc = 0; dc < 4; ++dc) {
            const unsigned short* vb =
                v_ws + ((size_t)bh * DH + dc * 16 + ln) * NKVP + kb;
            v16bf va = ld_frag8x2(vb + 16 * lh, vb + 8 + 16 * lh);
#pragma unroll
            for (int u = 0; u < 2; ++u)
                OT[u][dc] = wmma_bf(va, pf[u], OT[u][dc]);
        }
    }

#pragma unroll
    for (int u = 0; u < 2; ++u) {
        float invl = 1.0f / l_run[u];
#pragma unroll
        for (int dc = 0; dc < 4; ++dc)
#pragma unroll
            for (int r = 0; r < 8; ++r) {
                int d = dc * 16 + 8 * lh + r;
                float val = OT[u][dc][r] * invl;
                y_ws[((size_t)b * HW + qbase + u * 16 + ln) * HID + h * DH + d] = f2bf(val);
            }
    }
}

// ---------------- kernel 6: output projection (WMMA, 16x64/wave) -> fp32 --
__global__ void k_gemm_out(const unsigned short* __restrict__ wout_bf,
                           const unsigned short* __restrict__ y_ws,
                           float* __restrict__ out) {
    const int NB = HW / 64;              // 36
    const int TM = CH / 16;              // 16
    int bid = blockIdx.x;
    int nb  = bid % NB;
    int tm  = (bid / NB) % TM;
    int b   = bid / (NB * TM);
    int lane = threadIdx.x & 31;
    int ln = lane & 15, lh = lane >> 4;

    v8f acc[4] = {};
#pragma unroll
    for (int kc = 0; kc < HID / 32; ++kc) {
        const unsigned short* Ab = wout_bf + (size_t)(tm * 16 + ln) * HID + kc * 32;
        v16bf a = ld_frag8x2(Ab + 16 * lh, Ab + 8 + 16 * lh);
#pragma unroll
        for (int j = 0; j < 4; ++j) {
            const unsigned short* Bb =
                y_ws + ((size_t)b * HW + nb * 64 + j * 16 + ln) * HID + kc * 32 + 16 * lh;
            v16bf bm = ld_frag16(Bb);
            acc[j] = wmma_bf(a, bm, acc[j]);
        }
    }
#pragma unroll
    for (int j = 0; j < 4; ++j)
#pragma unroll
        for (int r = 0; r < 8; ++r) {
            int o = tm * 16 + 8 * lh + r;
            int p = nb * 64 + j * 16 + ln;
            out[((size_t)(b * CH + o)) * HW + p] = acc[j][r];
        }
}

// ---------------- host launcher -------------------------------------------
extern "C" void kernel_launch(void* const* d_in, const int* in_sizes, int n_in,
                              void* d_out, int out_size, void* d_ws, size_t ws_size,
                              hipStream_t stream) {
    (void)in_sizes; (void)n_in; (void)out_size; (void)ws_size;
    const float* x      = (const float*)d_in[0];   // (2,256,48,48)
    const float* gamma  = (const float*)d_in[1];   // (256,1,1)
    const float* mem_kv = (const float*)d_in[2];   // (2,8,4,64)
    const float* w_qkv  = (const float*)d_in[3];   // (1536,256)
    const float* w_out  = (const float*)d_in[4];   // (256,512)
    float* out = (float*)d_out;

    // workspace carve-up (bf16 = ushort); every block size is a multiple of 16
    // elements so 32B alignment of vector loads is preserved.
    unsigned short* ws = (unsigned short*)d_ws;
    const size_t XN_E   = (size_t)BATCH * HW * CH;            // 1,179,648
    const size_t WQKV_E = (size_t)3 * HID * CH;               //   393,216
    const size_t WOUT_E = (size_t)CH * HID;                   //   131,072
    const size_t Q_E    = (size_t)BATCH * NHEADS * HW * DH;   // 2,359,296
    const size_t K_E    = (size_t)BATCH * NHEADS * NKVP * DH; // 2,392,064
    const size_t V_E    = K_E;
    unsigned short* xn_ws   = ws;
    unsigned short* wqkv_bf = xn_ws + XN_E;
    unsigned short* wout_bf = wqkv_bf + WQKV_E;
    unsigned short* q_ws    = wout_bf + WOUT_E;
    unsigned short* k_ws    = q_ws + Q_E;
    unsigned short* v_ws    = k_ws + K_E;
    unsigned short* y_ws    = v_ws + V_E;

    k_convert_weights<<<(524288 + 255) / 256, 256, 0, stream>>>(w_qkv, w_out, wqkv_bf, wout_bf);
    k_prep_kv<<<(32768 + 255) / 256, 256, 0, stream>>>(mem_kv, k_ws, v_ws);
    k_rmsnorm<<<BATCH * HW, CH, 0, stream>>>(x, gamma, xn_ws);
    k_gemm_qkv<<<BATCH * 96 * 36, 32, 0, stream>>>(wqkv_bf, xn_ws, q_ws, k_ws, v_ws);
    k_attention<<<BATCH * NHEADS * 72, 32, 0, stream>>>(q_ws, k_ws, v_ws, y_ws);
    k_gemm_out<<<BATCH * 16 * 36, 32, 0, stream>>>(wout_bf, y_ws, out);
}
